// SwinBiMamba_74466142978816
// MI455X (gfx1250) — compile-verified, matrix operand order checked by reading, not run on hardware
//
#include <hip/hip_runtime.h>
#include <hip/hip_bf16.h>
#include <math.h>

// ---------------------------------------------------------------------------
// SwinBiMamba for MI455X (gfx1250, wave32, WMMA).
// GEMMs: v_wmma_f32_16x16x32_f16, f16 operands / f32 accum.
// A panels staged into LDS with GLOBAL_LOAD_ASYNC_TO_LDS_B128 (ASYNCcnt path)
// so the VGPR file and load-return bus stay free for the B stream + WMMA.
// ---------------------------------------------------------------------------

#define BATCH_ 64
#define L_     256
#define D_     512
#define DI_    1024
#define DS_    16
#define DTR_   32
#define DC_    4
#define BL_    (BATCH_ * L_)   // 16384

typedef __attribute__((ext_vector_type(16))) _Float16 v16h;
typedef __attribute__((ext_vector_type(8)))  _Float16 v8h;
typedef __attribute__((ext_vector_type(8)))  float    v8f;

// ---------------------------------------------------------------------------
// fp32 -> fp16 weight conversion (once per launch; weights then live in L2)
// ---------------------------------------------------------------------------
__global__ void cvt_f32_f16(const float* __restrict__ src,
                            _Float16* __restrict__ dst, int n) {
    int i = blockIdx.x * 256 + threadIdx.x;
    if (i < n) dst[i] = (_Float16)src[i];
}

// ---------------------------------------------------------------------------
// Fused permute + residual add + RMSNorm.  One block per (b, l) row of D=512.
// ---------------------------------------------------------------------------
__global__ __launch_bounds__(128)
void permute_add_norm(const float* __restrict__ hsrc,
                      const float* __restrict__ rsrc,
                      const int* __restrict__ perm,
                      const float* __restrict__ nw,
                      float* __restrict__ rdst,
                      _Float16* __restrict__ ndst) {
    int blk = blockIdx.x;
    int b = blk >> 8;          // / L_
    int l = blk & (L_ - 1);
    int sl = perm[l];
    size_t srow = (size_t)(b * L_ + sl) * D_;
    const float4* hp = (const float4*)(hsrc + srow);
    float4 h = hp[threadIdx.x];
    if (rsrc) {
        const float4* rp = (const float4*)(rsrc + srow);
        float4 r = rp[threadIdx.x];
        h.x += r.x; h.y += r.y; h.z += r.z; h.w += r.w;
    }
    float ss = h.x * h.x + h.y * h.y + h.z * h.z + h.w * h.w;
    #pragma unroll
    for (int off = 16; off > 0; off >>= 1) ss += __shfl_xor(ss, off, 32);
    __shared__ float red[4];
    int wid = threadIdx.x >> 5;
    if ((threadIdx.x & 31) == 0) red[wid] = ss;
    __syncthreads();
    float tot = red[0] + red[1] + red[2] + red[3];
    float inv = rsqrtf(tot * (1.0f / (float)D_) + 1e-5f);

    size_t drow = (size_t)blk * D_;
    ((float4*)(rdst + drow))[threadIdx.x] = h;
    int c = threadIdx.x * 4;
    _Float16* np = ndst + drow + c;
    np[0] = (_Float16)(h.x * inv * nw[c + 0]);
    np[1] = (_Float16)(h.y * inv * nw[c + 1]);
    np[2] = (_Float16)(h.z * inv * nw[c + 2]);
    np[3] = (_Float16)(h.w * inv * nw[c + 3]);
}

// ---------------------------------------------------------------------------
// Final RMSNorm over (hidden + residual) -> f32 output.
// ---------------------------------------------------------------------------
__global__ __launch_bounds__(128)
void final_add_norm(const float* __restrict__ hsrc,
                    const float* __restrict__ rsrc,
                    const float* __restrict__ nw,
                    float* __restrict__ out) {
    size_t row = (size_t)blockIdx.x * D_;
    const float4* hp = (const float4*)(hsrc + row);
    const float4* rp = (const float4*)(rsrc + row);
    float4 h = hp[threadIdx.x];
    float4 r = rp[threadIdx.x];
    h.x += r.x; h.y += r.y; h.z += r.z; h.w += r.w;
    float ss = h.x * h.x + h.y * h.y + h.z * h.z + h.w * h.w;
    #pragma unroll
    for (int off = 16; off > 0; off >>= 1) ss += __shfl_xor(ss, off, 32);
    __shared__ float red[4];
    int wid = threadIdx.x >> 5;
    if ((threadIdx.x & 31) == 0) red[wid] = ss;
    __syncthreads();
    float tot = red[0] + red[1] + red[2] + red[3];
    float inv = rsqrtf(tot * (1.0f / (float)D_) + 1e-5f);
    int c = threadIdx.x * 4;
    float4 o;
    o.x = h.x * inv * nw[c + 0];
    o.y = h.y * inv * nw[c + 1];
    o.z = h.z * inv * nw[c + 2];
    o.w = h.w * inv * nw[c + 3];
    ((float4*)(out + row))[threadIdx.x] = o;
}

// ---------------------------------------------------------------------------
// WMMA GEMM:  C[M,N] = A[M,K] * W[N,K]^T    (einsum 'mk,nk->mn')
// Block = 256 threads = 8 waves.  Block tile 128(M) x 64(N); each wave owns a
// 16x64 strip (4 accumulators).  A panel (128x32 f16, 8KB) staged into LDS
// via GLOBAL_LOAD_ASYNC_TO_LDS_B128 (2 x 16B chunks per thread, ASYNCcnt).
// Fragment gathers follow the documented 16-bit A/B VGPR layout:
//   lane[0..15]  row=lane,    halves K[0..7]  and K[16..23]
//   lane[16..31] row=lane-16, halves K[8..15] and K[24..31]
// epilogue: 0 = none, 1 = softplus(acc + bias[n])
// store_f16: 1 -> f16 C, 0 -> f32 C
// ---------------------------------------------------------------------------
__device__ inline v16h frag_gather(const _Float16* p, int koff) {
    const v8h* p0 = (const v8h*)(p + koff);
    const v8h* p1 = (const v8h*)(p + koff + 16);
    v8h x0 = p0[0];
    v8h x1 = p1[0];
    v16h out;
    #pragma unroll
    for (int i = 0; i < 8; ++i) { out[i] = x0[i]; out[i + 8] = x1[i]; }
    return out;
}

__global__ __launch_bounds__(256)
void gemm_wmma(const _Float16* __restrict__ A, int lda,
               const _Float16* __restrict__ W,
               void* __restrict__ Cout,
               int M, int N, int K,
               int store_f16, int epilogue,
               const float* __restrict__ bias) {
    __shared__ __align__(16) _Float16 sA[128 * 32];
    int m0 = blockIdx.x * 128;
    int n0 = blockIdx.y * 64;
    int tid = threadIdx.x;
    int lane = tid & 31;
    int wid = tid >> 5;

    // Generic LDS pointer low 32 bits == LDS byte offset (ISA 10.2 aperture).
    unsigned sA_base = (unsigned)(size_t)(&sA[0]);

    v8f acc[4];
    #pragma unroll
    for (int f = 0; f < 4; ++f)
        #pragma unroll
        for (int e = 0; e < 8; ++e) acc[f][e] = 0.0f;

    int r = lane & 15;
    int koff = (lane >> 4) << 3;          // 0 or 8

    for (int kb = 0; kb < K; kb += 32) {
        // ---- async DMA stage of A panel: 128 rows x 32 halves = 8KB ----
        // 512 x 16B chunks; each of 256 threads issues 2 async b128 copies
        // directly into LDS (no VGPR round-trip, tracked by ASYNCcnt).
        {
            #pragma unroll
            for (int cc = 0; cc < 2; ++cc) {
                unsigned c = (unsigned)tid + (unsigned)cc * 256u;
                unsigned lds_off = sA_base + c * 16u;
                const _Float16* gp =
                    A + (size_t)(m0 + (int)(c >> 2)) * lda + kb + (int)(c & 3u) * 8;
                unsigned long long ga = (unsigned long long)(size_t)gp;
                asm volatile(
                    "global_load_async_to_lds_b128 %0, %1, off"
                    :
                    : "v"(lds_off), "v"(ga)
                    : "memory");
            }
            if (kb + 32 < K)   // pull next A panel toward the WGP (prefetch)
                __builtin_prefetch(A + (size_t)(m0 + (tid >> 1)) * lda + kb + 32, 0, 3);
        }
        // wave's own async copies complete, then block barrier for the panel
        asm volatile("s_wait_asynccnt 0x0" ::: "memory");
        __syncthreads();

        v16h a = frag_gather(sA + (wid * 16 + r) * 32, koff);
        #pragma unroll
        for (int f = 0; f < 4; ++f) {
            v16h bfr = frag_gather(W + (size_t)(n0 + f * 16 + r) * K + kb, koff);
            acc[f] = __builtin_amdgcn_wmma_f32_16x16x32_f16(
                false, a, false, bfr, (short)0, acc[f], false, false);
        }
        __syncthreads();
    }

    // ---- epilogue / store ----
    int rowb = (lane >> 4) * 8;
    int col  = lane & 15;
    #pragma unroll
    for (int f = 0; f < 4; ++f) {
        int gn = n0 + f * 16 + col;
        float bv = (epilogue == 1) ? bias[gn] : 0.0f;
        #pragma unroll
        for (int rr = 0; rr < 8; ++rr) {
            int gm = m0 + wid * 16 + rowb + rr;
            float v = acc[f][rr];
            if (epilogue == 1) {
                v += bv;
                v = (v > 20.0f) ? v : log1pf(__expf(v));   // softplus
            }
            size_t o = (size_t)gm * N + gn;
            if (store_f16) ((_Float16*)Cout)[o] = (_Float16)v;
            else           ((float*)Cout)[o]    = v;
        }
    }
}

// ---------------------------------------------------------------------------
// Causal depthwise conv (width 4) + SiLU.  rev=1 reads the flipped sequence.
// xz holds [xm | z] with row width 2*DI; uc is BLxDI in scan order.
// ---------------------------------------------------------------------------
__global__ __launch_bounds__(256)
void conv_silu(const _Float16* __restrict__ xz,
               const float* __restrict__ cw,
               const float* __restrict__ cb,
               _Float16* __restrict__ uc, int rev) {
    int idx = blockIdx.x * 256 + threadIdx.x;       // over BL*DI
    int ch = idx & (DI_ - 1);
    int bl = idx >> 10;
    int b = bl >> 8;
    int l = bl & (L_ - 1);
    float acc = cb[ch];
    #pragma unroll
    for (int j = 0; j < DC_; ++j) {
        int lx = l - (DC_ - 1) + j;
        if (lx >= 0) {
            int srcl = rev ? (L_ - 1 - lx) : lx;
            float xv = (float)xz[(size_t)(b * L_ + srcl) * (2 * DI_) + ch];
            acc += cw[ch * DC_ + j] * xv;
        }
    }
    float s = acc / (1.0f + __expf(-acc));           // silu
    uc[(size_t)bl * DI_ + ch] = (_Float16)s;
}

// ---------------------------------------------------------------------------
// Selective scan.  One lane per (b, di): 16-element state kept in VGPRs,
// sequential over L=256.  65536 lanes = 2048 wave32 recurrences in flight.
// rev=1: scan over flipped time, write y back at flipped index, use m_rev.
// accum=1: y += (second direction).
// ---------------------------------------------------------------------------
__global__ __launch_bounds__(256)
void scan_kernel(const _Float16* __restrict__ uc,
                 const _Float16* __restrict__ dt16,
                 const _Float16* __restrict__ xdb,
                 const float* __restrict__ mask,
                 const float* __restrict__ A_log,
                 const float* __restrict__ Dp,
                 float* __restrict__ y,
                 int rev, int accum) {
    int idx = blockIdx.x * 256 + threadIdx.x;       // over B*DI
    int di = idx & (DI_ - 1);
    int b = idx >> 10;

    float A[DS_];
    #pragma unroll
    for (int s = 0; s < DS_; ++s) A[s] = -__expf(A_log[di * DS_ + s]);
    float Dv = Dp[di];

    float h[DS_];
    #pragma unroll
    for (int s = 0; s < DS_; ++s) h[s] = 0.0f;

    for (int t = 0; t < L_; ++t) {
        size_t row = (size_t)b * L_ + t;
        float dtv = (float)dt16[row * DI_ + di];
        float uv  = (float)uc[row * DI_ + di];
        float m = rev ? mask[(L_ - t) & (L_ - 1)] : mask[t];
        const _Float16* bc = xdb + row * 64;
        v8h Bv0 = *(const v8h*)(bc + 32);
        v8h Bv1 = *(const v8h*)(bc + 40);
        v8h Cv0 = *(const v8h*)(bc + 48);
        v8h Cv1 = *(const v8h*)(bc + 56);
        float du = dtv * uv;
        float yt = 0.0f;
        #pragma unroll
        for (int s = 0; s < DS_; ++s) {
            float Bs = (s < 8) ? (float)Bv0[s & 7] : (float)Bv1[s & 7];
            float Cs = (s < 8) ? (float)Cv0[s & 7] : (float)Cv1[s & 7];
            float dA = __expf(dtv * A[s]) * m;
            h[s] = dA * h[s] + du * Bs;
            yt += h[s] * Cs;
        }
        yt += uv * Dv;
        int lo = rev ? (L_ - 1 - t) : t;
        size_t orow = ((size_t)b * L_ + lo) * DI_ + di;
        if (accum) y[orow] += yt; else y[orow] = yt;
    }
}

// ---------------------------------------------------------------------------
// Gate: ymul = (y_f + y_b) * silu(z)  -> f16 (A operand of W_out GEMM)
// ---------------------------------------------------------------------------
__global__ __launch_bounds__(256)
void gate_mul(const float* __restrict__ y,
              const _Float16* __restrict__ xz,
              _Float16* __restrict__ ym) {
    int idx = blockIdx.x * 256 + threadIdx.x;       // over BL*DI
    int di = idx & (DI_ - 1);
    int bl = idx >> 10;
    float z = (float)xz[(size_t)bl * (2 * DI_) + DI_ + di];
    float g = z / (1.0f + __expf(-z));
    ym[idx] = (_Float16)(y[idx] * g);
}

// ---------------------------------------------------------------------------
// Host-side orchestration (all on `stream`, graph-capture safe).
// ---------------------------------------------------------------------------
extern "C" void kernel_launch(void* const* d_in, const int* in_sizes, int n_in,
                              void* d_out, int out_size, void* d_ws, size_t ws_size,
                              hipStream_t stream) {
    const float* x      = (const float*)d_in[0];
    const float* maskA[2] = {(const float*)d_in[1], (const float*)d_in[2]};
    const int*   permA[2] = {(const int*)d_in[3], (const int*)d_in[4]};
    const float* W_in   = (const float*)d_in[5];
    const float* W_out  = (const float*)d_in[6];
    const float* conv_w = (const float*)d_in[7];
    const float* conv_b = (const float*)d_in[8];
    const float* Wx     = (const float*)d_in[9];
    const float* Wdt    = (const float*)d_in[10];
    const float* bdt    = (const float*)d_in[11];
    const float* A_log  = (const float*)d_in[12];
    const float* Dp     = (const float*)d_in[13];
    const float* norm_w = (const float*)d_in[14];
    const float* fnw    = (const float*)d_in[15];

    // ---- carve workspace ----
    char* p = (char*)d_ws;
    auto carve = [&](size_t bytes) {
        char* q = p;
        p += (bytes + 255) & ~(size_t)255;
        return (void*)q;
    };
    float*    r_a    = (float*)   carve((size_t)BL_ * D_ * 4);
    float*    r_b    = (float*)   carve((size_t)BL_ * D_ * 4);
    _Float16* n16    = (_Float16*)carve((size_t)BL_ * D_ * 2);
    _Float16* xz16   = (_Float16*)carve((size_t)BL_ * 2 * DI_ * 2);
    _Float16* uc16   = (_Float16*)carve((size_t)BL_ * DI_ * 2);
    _Float16* xdb16  = (_Float16*)carve((size_t)BL_ * 64 * 2);
    _Float16* dt16   = (_Float16*)carve((size_t)BL_ * DI_ * 2);
    float*    y32    = (float*)   carve((size_t)BL_ * DI_ * 4);
    _Float16* ymul16 = (_Float16*)carve((size_t)BL_ * DI_ * 2);
    float*    h32    = (float*)   carve((size_t)BL_ * D_ * 4);
    _Float16* Win16  = (_Float16*)carve((size_t)2 * 2 * DI_ * D_ * 2);
    _Float16* Wout16 = (_Float16*)carve((size_t)2 * D_ * DI_ * 2);
    _Float16* Wx16   = (_Float16*)carve((size_t)2 * 2 * 64 * DI_ * 2);
    _Float16* Wdt16  = (_Float16*)carve((size_t)2 * 2 * DI_ * DTR_ * 2);
    (void)ws_size; (void)n_in; (void)in_sizes; (void)out_size;

    // ---- convert weights to f16 once (tiny; lives in L2 afterwards) ----
    {
        int n1 = 2 * 2 * DI_ * D_;      // W_in: 2,097,152
        int n2 = 2 * D_ * DI_;          // W_out: 1,048,576
        int n3 = 2 * 2 * 64 * DI_;      // Wx: 262,144
        int n4 = 2 * 2 * DI_ * DTR_;    // Wdt: 131,072
        cvt_f32_f16<<<(n1 + 255) / 256, 256, 0, stream>>>(W_in,  Win16,  n1);
        cvt_f32_f16<<<(n2 + 255) / 256, 256, 0, stream>>>(W_out, Wout16, n2);
        cvt_f32_f16<<<(n3 + 255) / 256, 256, 0, stream>>>(Wx,    Wx16,   n3);
        cvt_f32_f16<<<(n4 + 255) / 256, 256, 0, stream>>>(Wdt,   Wdt16,  n4);
    }

    float* res_bufs[2] = {r_a, r_b};
    const float* hsrc = x;
    const float* rprev = nullptr;
    float* rcur = nullptr;

    for (int i = 0; i < 2; ++i) {
        rcur = res_bufs[i];
        // permute + residual + rmsnorm -> rcur (f32), n16 (f16)
        permute_add_norm<<<BL_, 128, 0, stream>>>(
            hsrc, rprev, permA[i], norm_w + i * D_, rcur, n16);

        // xz = normed @ W_in^T   (M=16384, N=2048, K=512) -> f16
        gemm_wmma<<<dim3(BL_ / 128, 2048 / 64), 256, 0, stream>>>(
            n16, D_, Win16 + (size_t)i * 2 * DI_ * D_, xz16,
            BL_, 2 * DI_, D_, 1, 0, nullptr);

        for (int d = 0; d < 2; ++d) {
            int sd = i * 2 + d;
            // uc = silu(causal_dwconv(xm_dir))
            conv_silu<<<(BL_ * DI_) / 256, 256, 0, stream>>>(
                xz16, conv_w + (size_t)sd * DI_ * DC_, conv_b + (size_t)sd * DI_,
                uc16, d);
            // xdb = uc @ Wx^T      (N=64, K=1024) -> f16
            gemm_wmma<<<dim3(BL_ / 128, 64 / 64), 256, 0, stream>>>(
                uc16, DI_, Wx16 + (size_t)sd * 64 * DI_, xdb16,
                BL_, 64, DI_, 1, 0, nullptr);
            // dt = softplus(dtr @ Wdt^T + bdt)  (N=1024, K=32) -> f16
            gemm_wmma<<<dim3(BL_ / 128, DI_ / 64), 256, 0, stream>>>(
                xdb16, 64, Wdt16 + (size_t)sd * DI_ * DTR_, dt16,
                BL_, DI_, DTR_, 1, 1, bdt + (size_t)sd * DI_);
            // selective scan (d==1 accumulates into y32)
            scan_kernel<<<(BATCH_ * DI_) / 256, 256, 0, stream>>>(
                uc16, dt16, xdb16, maskA[i],
                A_log + (size_t)sd * DI_ * DS_, Dp + (size_t)sd * DI_,
                y32, d, d);
        }

        // ymul = (y_f + y_b) * silu(z) -> f16
        gate_mul<<<(BL_ * DI_) / 256, 256, 0, stream>>>(y32, xz16, ymul16);

        // hidden = ymul @ W_out^T  (N=512, K=1024) -> f32
        gemm_wmma<<<dim3(BL_ / 128, D_ / 64), 256, 0, stream>>>(
            ymul16, DI_, Wout16 + (size_t)i * D_ * DI_, h32,
            BL_, D_, DI_, 0, 0, nullptr);

        hsrc = h32;
        rprev = rcur;
    }

    // out = rmsnorm(hidden + residual, final_norm_w)
    final_add_norm<<<BL_, 128, 0, stream>>>(h32, rprev, fnw, (float*)d_out);
}
